// IntervenedGNN_50757923504434
// MI455X (gfx1250) — matrix operand order repaired and checked by reading, not compiled, not run on hardware
//
#include <hip/hip_runtime.h>

typedef __attribute__((ext_vector_type(16))) _Float16 v16h;
typedef __attribute__((ext_vector_type(8)))  _Float16 v8h;
typedef __attribute__((ext_vector_type(4)))  _Float16 v4h;
typedef __attribute__((ext_vector_type(8)))  float    v8f;
typedef int v4i_g __attribute__((vector_size(16)));  // matches async builtin param type

#define IN_DIM 256
#define HID    512
#define OUT_DIM 128

// ---- CDNA5 async global->LDS copy (ASYNCcnt path), with portable fallback --
#if defined(__has_builtin)
#if __has_builtin(__builtin_amdgcn_global_load_async_to_lds_b128)
#define HAS_ASYNC_LDS 1
#endif
#endif
#ifndef HAS_ASYNC_LDS
#define HAS_ASYNC_LDS 0
#endif

#if HAS_ASYNC_LDS
__device__ __forceinline__ void async_copy16(const _Float16* g, _Float16* l) {
    __builtin_amdgcn_global_load_async_to_lds_b128(
        (__attribute__((address_space(1))) v4i_g*)(g),
        (__attribute__((address_space(3))) v4i_g*)(l), 0, 0);
}
#endif

__device__ __forceinline__ void async_wait0() {
#if HAS_ASYNC_LDS
    asm volatile("s_wait_asynccnt 0x0" ::: "memory");
#endif
}

// ---------------------------------------------------------------------------
// row_ptr[r] = lower_bound(adj_row, r) over the SORTED row array
// ---------------------------------------------------------------------------
__global__ void build_row_ptr(const int* __restrict__ rows, long long E, int N,
                              int* __restrict__ row_ptr) {
    int r = blockIdx.x * blockDim.x + threadIdx.x;
    if (r > N) return;
    long long lo = 0, hi = E;
    while (lo < hi) {
        long long mid = (lo + hi) >> 1;
        if (rows[mid] < r) lo = mid + 1; else hi = mid;
    }
    row_ptr[r] = (int)lo;
}

// ---------------------------------------------------------------------------
// Weight conversion: f32 [K][N] row-major -> f16 [N][K] (n-major) so GEMM
// B tiles are contiguous-in-K 16B chunks (async-copyable, no LDS transpose).
// ---------------------------------------------------------------------------
__global__ void convert_transpose_w(const float* __restrict__ src,
                                    _Float16* __restrict__ dst, int Kd, int Nd) {
    int i = blockIdx.x * blockDim.x + threadIdx.x;
    if (i >= Kd * Nd) return;
    int k = i / Nd, n = i % Nd;
    dst[(size_t)n * Kd + k] = (_Float16)src[i];
}

// ---------------------------------------------------------------------------
// CSR SpMM: one wave per row, f32 register accumulation, NO atomics.
// Lanes stripe the feature dim (coalesced 16B/8B gathers); col/val are
// wave-uniform broadcast loads. Output stored directly as f16.
// ---------------------------------------------------------------------------
__device__ __forceinline__ float4 load4f(const float* p) {
    return *(const float4*)p;
}
__device__ __forceinline__ float4 load4f(const _Float16* p) {
    v4h h = *(const v4h*)p;
    return make_float4((float)h[0], (float)h[1], (float)h[2], (float)h[3]);
}

template <int D, typename SrcT>
__global__ __launch_bounds__(256)
void spmm_csr(const int* __restrict__ row_ptr, const int* __restrict__ col,
              const float* __restrict__ val, const SrcT* __restrict__ h,
              _Float16* __restrict__ out, int N) {
    constexpr int C = D / 128;  // float4 chunks per lane (2 for 256, 4 for 512)
    const int wave = blockIdx.x * (blockDim.x >> 5) + (threadIdx.x >> 5);
    const int lane = threadIdx.x & 31;
    if (wave >= N) return;

    const int start = row_ptr[wave];
    const int end   = row_ptr[wave + 1];

    float acc[C * 4] = {};
    for (int e = start; e < end; ++e) {
        const int   c = col[e];     // wave-uniform
        const float v = val[e];     // wave-uniform
        const SrcT* hp = h + (size_t)c * D + lane * 4;
#pragma unroll
        for (int i = 0; i < C; ++i) {
            float4 f = load4f(hp + i * 128);
            acc[i * 4 + 0] += v * f.x;
            acc[i * 4 + 1] += v * f.y;
            acc[i * 4 + 2] += v * f.z;
            acc[i * 4 + 3] += v * f.w;
        }
    }

    _Float16* op = out + (size_t)wave * D + lane * 4;
#pragma unroll
    for (int i = 0; i < C; ++i) {
        v4h o = {(_Float16)acc[i * 4 + 0], (_Float16)acc[i * 4 + 1],
                 (_Float16)acc[i * 4 + 2], (_Float16)acc[i * 4 + 3]};
        *(v4h*)(op + i * 128) = o;
    }
}

// ---------------------------------------------------------------------------
// Tile staging: TR rows x 32 halves (64B/row), 256 threads, 16B chunks
// ---------------------------------------------------------------------------
template <int TR>
__device__ __forceinline__ void stage_manual(const _Float16* __restrict__ src, size_t ld,
                                             int rows_avail, _Float16 (*dst)[40], int t) {
#pragma unroll
    for (int i = 0; i < (TR * 4) / 256; ++i) {
        int q = t + i * 256;
        int r = q >> 2;
        int c8 = (q & 3) * 8;
        uint4 v = {0u, 0u, 0u, 0u};
        if (r < rows_avail) v = *(const uint4*)(src + (size_t)r * ld + c8);
        *(uint4*)&dst[r][c8] = v;
    }
}

#if HAS_ASYNC_LDS
template <int TR>
__device__ __forceinline__ void stage_async(const _Float16* __restrict__ src, size_t ld,
                                            _Float16 (*dst)[40], int t) {
#pragma unroll
    for (int i = 0; i < (TR * 4) / 256; ++i) {
        int q = t + i * 256;
        int r = q >> 2;
        int c8 = (q & 3) * 8;
        async_copy16(src + (size_t)r * ld + c8, &dst[r][c8]);
    }
}
#endif

// ---------------------------------------------------------------------------
// WMMA GEMM: out[M,NOUT] = act(A[M,K] @ W[K,NOUT] + bias)
//  - A: f16 row-major [M][K]; Wt: f16 n-major [NOUT][K]
//  - 256 threads = 8 waves; block tile 128(M) x 128(N)
//  - double-buffered LDS, async global->LDS staging on full tiles
//  - ZERO4 fuses the Intervention (zero out channels 0..3)
// ---------------------------------------------------------------------------
template <int K, int NOUT, bool RELU, bool ZERO4, typename OutT>
__global__ __launch_bounds__(256)
void gemm_wmma(const _Float16* __restrict__ A, const _Float16* __restrict__ Wt,
               const float* __restrict__ bias, OutT* __restrict__ out, int M) {
    constexpr int TM = 128, TN = 128;
    constexpr int KSTEPS = K / 32;
    __shared__ _Float16 Asl[2][TM][40];  // 80B rows -> 16B-aligned chunks
    __shared__ _Float16 Bsl[2][TN][40];

    const int tileN = blockIdx.x * TN;
    const int tileM = blockIdx.y * TM;
    const int t    = threadIdx.x;
    const int w    = t >> 5;
    const int lane = t & 31;
    const int half = lane >> 4;
    const int l16  = lane & 15;
    const bool full = (tileM + TM) <= M;

    const _Float16* Abase = A + (size_t)tileM * K;   // [r][k]
    const _Float16* Bbase = Wt + (size_t)tileN * K;  // [n][k]

    v8f acc[8] = {};

    auto stage = [&](int p, int k0) {
#if HAS_ASYNC_LDS
        if (full) {
            stage_async<TM>(Abase + k0, K, Asl[p], t);
            stage_async<TN>(Bbase + k0, K, Bsl[p], t);
            return;
        }
#endif
        stage_manual<TM>(Abase + k0, K, M - tileM, Asl[p], t);
        stage_manual<TN>(Bbase + k0, K, TN, Bsl[p], t);
    };

    stage(0, 0);
    async_wait0();
    __syncthreads();

    for (int s = 0; s < KSTEPS; ++s) {
        const int p = s & 1;
        if (s + 1 < KSTEPS) stage(1 - p, (s + 1) * 32);

        // A fragment: lanes 0-15 K{0-7,16-23}, lanes 16-31 K{8-15,24-31}
        const int m = w * 16 + l16;
        const _Float16* ap = &Asl[p][m][half * 8];
        v8h alo = *(const v8h*)(ap);
        v8h ahi = *(const v8h*)(ap + 16);
        v16h a = __builtin_shufflevector(alo, ahi, 0, 1, 2, 3, 4, 5, 6, 7,
                                         8, 9, 10, 11, 12, 13, 14, 15);
#pragma unroll
        for (int j = 0; j < 8; ++j) {
            // B fragment: lane n = j*16+l16, lanes 0-15 K{0-7,16-23}, lanes 16-31 K{8-15,24-31}
            const _Float16* bp = &Bsl[p][j * 16 + l16][half * 8];
            v8h blo = *(const v8h*)(bp);
            v8h bhi = *(const v8h*)(bp + 16);
            v16h b = __builtin_shufflevector(blo, bhi, 0, 1, 2, 3, 4, 5, 6, 7,
                                             8, 9, 10, 11, 12, 13, 14, 15);
            acc[j] = __builtin_amdgcn_wmma_f32_16x16x32_f16(
                false, a, false, b, (short)0, acc[j], false, false);
        }
        async_wait0();
        __syncthreads();
    }

    // epilogue: bias + ReLU (+ Intervention), store f16 or f32
#pragma unroll
    for (int j = 0; j < 8; ++j) {
        int gn = tileN + j * 16 + l16;
        float bb = bias[gn];
#pragma unroll
        for (int i = 0; i < 8; ++i) {
            int gm = tileM + w * 16 + i + half * 8;  // C layout: VGPR i -> M=i(+8)
            if (gm < M) {
                float v = acc[j][i] + bb;
                if (RELU) v = fmaxf(v, 0.f);
                if (ZERO4 && gn < 4) v = 0.f;
                out[(size_t)gm * NOUT + gn] = (OutT)v;
            }
        }
    }
}

// ---------------------------------------------------------------------------
// Host-side orchestration
// ---------------------------------------------------------------------------
extern "C" void kernel_launch(void* const* d_in, const int* in_sizes, int n_in,
                              void* d_out, int out_size, void* d_ws, size_t ws_size,
                              hipStream_t stream) {
    const float* x    = (const float*)d_in[0];
    const float* aval = (const float*)d_in[1];
    const float* W1   = (const float*)d_in[2];
    const float* b1   = (const float*)d_in[3];
    const float* W2   = (const float*)d_in[4];
    const float* b2   = (const float*)d_in[5];
    const float* Wf   = (const float*)d_in[6];
    const float* bf   = (const float*)d_in[7];
    const int*   arow = (const int*)d_in[8];
    const int*   acol = (const int*)d_in[9];
    float* out = (float*)d_out;

    const int N = in_sizes[0] / IN_DIM;
    const long long E = in_sizes[1];

    // workspace layout (all f16 activations; no f32 agg buffer, no memsets)
    _Float16* bufC = (_Float16*)d_ws;        // agg1h [N,256] then agg2h [N,512]
    _Float16* bufD = bufC + (size_t)N * HID; // h1 / h2 [N,512]
    _Float16* W1t  = bufD + (size_t)N * HID; // [HID][IN_DIM]
    _Float16* W2t  = W1t + IN_DIM * HID;     // [HID][HID]
    _Float16* Wft  = W2t + HID * HID;        // [OUT_DIM][HID]
    int* row_ptr   = (int*)(Wft + HID * OUT_DIM); // [N+1]

    // 1) CSR row pointers from the sorted row array
    build_row_ptr<<<(N + 1 + 255) / 256, 256, 0, stream>>>(arow, E, N, row_ptr);

    // 2) weights: convert + transpose to n-major f16 (once, tiny)
    convert_transpose_w<<<(IN_DIM * HID + 255) / 256, 256, 0, stream>>>(W1, W1t, IN_DIM, HID);
    convert_transpose_w<<<(HID * HID + 255) / 256, 256, 0, stream>>>(W2, W2t, HID, HID);
    convert_transpose_w<<<(HID * OUT_DIM + 255) / 256, 256, 0, stream>>>(Wf, Wft, HID, OUT_DIM);

    // 3) SpMM layer 1: agg1 = A * x (f32 gather, reg accumulate, f16 store)
    {
        int blocks = (N + 7) / 8;  // 8 waves per block, 1 row per wave
        spmm_csr<IN_DIM, float><<<blocks, 256, 0, stream>>>(row_ptr, acol, aval, x, bufC, N);
    }

    // 4) h1 = relu(agg1 @ W1 + b1) -> f16
    {
        dim3 grid(HID / 128, (N + 127) / 128);
        gemm_wmma<IN_DIM, HID, true, false, _Float16><<<grid, 256, 0, stream>>>(bufC, W1t, b1, bufD, N);
    }

    // 5) SpMM layer 2: agg2 = A * h1 (f16 gather, reg accumulate, f16 store)
    {
        int blocks = (N + 7) / 8;
        spmm_csr<HID, _Float16><<<blocks, 256, 0, stream>>>(row_ptr, acol, aval, bufD, bufC, N);
    }

    // 6) h2 = relu(agg2 @ W2 + b2), Intervention fused (zero ch 0..3) -> f16
    {
        dim3 grid(HID / 128, (N + 127) / 128);
        gemm_wmma<HID, HID, true, true, _Float16><<<grid, 256, 0, stream>>>(bufC, W2t, b2, bufD, N);
    }

    // 7) out = h_int @ Wf + bf (f32 output)
    {
        dim3 grid(OUT_DIM / 128, (N + 127) / 128);
        gemm_wmma<HID, OUT_DIM, false, false, float><<<grid, 256, 0, stream>>>(bufD, Wft, bf, out, N);
    }
}
